// BahdanauAttention_28879360098342
// MI455X (gfx1250) — matrix operand order repaired and checked by reading, not compiled, not run on hardware
//
#include <hip/hip_runtime.h>
#include <hip/hip_bf16.h>

// Sizes fixed by the reference.
#define B_  64
#define S_  1024
#define F_  1024
#define U_  1024

typedef __bf16 bf16_t;
typedef bf16_t v16bf __attribute__((ext_vector_type(16)));
typedef float  v8f   __attribute__((ext_vector_type(8)));

union FragU {
    struct { uint4 lo, hi; } q;   // 32 bytes = 8 VGPRs of packed bf16
    v16bf v;
};

__device__ __forceinline__ unsigned short f32_to_bf16_rne(float f) {
    unsigned int u = __float_as_uint(f);
    unsigned int r = u + 0x7FFFu + ((u >> 16) & 1u);   // round-to-nearest-even
    return (unsigned short)(r >> 16);
}

__device__ __forceinline__ unsigned int pack_bf16x2(float lo, float hi) {
    return (unsigned int)f32_to_bf16_rne(lo) |
           ((unsigned int)f32_to_bf16_rne(hi) << 16);
}

// CDNA5 hardware TANH transcendental when exposed; branch-free v_exp_f32 fallback.
__device__ __forceinline__ float fast_tanh(float x) {
#if __has_builtin(__builtin_amdgcn_tanhf)
    return __builtin_amdgcn_tanhf(x);
#else
    float ax = __builtin_fabsf(x);
    float e  = __expf(ax + ax);
    float t  = 1.f - 2.f / (e + 1.f);
    return __builtin_copysignf(t, x);
#endif
}

// ---------------------------------------------------------------------------
// Kernel A: proj_h[b,u] = hidden[b,:] . W2[:,u] + W2_b[u]
// ---------------------------------------------------------------------------
__global__ __launch_bounds__(256) void proj_hidden_kernel(
    const float* __restrict__ hidden, const float* __restrict__ W2,
    const float* __restrict__ W2b, float* __restrict__ projh)
{
    const int u = blockIdx.x * 256 + threadIdx.x;
    const int b = blockIdx.y;
    const float* hrow = hidden + b * U_;
    float acc = W2b[u];
#pragma unroll 8
    for (int k = 0; k < U_; ++k)
        acc += hrow[k] * W2[(size_t)k * U_ + u];
    projh[b * U_ + u] = acc;
}

// ---------------------------------------------------------------------------
// Kernel B: fused  logits[r] = V . tanh(features[r,:] @ W1 + W1_b + proj_h[b])
// Dynamic LDS: A panel 128x1024 bf16 (256 KB, staged ONCE) +
//              double-buffered W1 tile 2 x (128n x 32k) bf16 (16 KB).
// Wave partition: 8 waves = 4 row-quarters x 2 n-halves. Each wave owns
// 32 rows (2 A frags) x 64 cols (4 B frags): every B fragment feeds TWO
// back-to-back WMMAs, doubling matrix-pipe duty per LDS round-trip.
// ---------------------------------------------------------------------------
__global__ __launch_bounds__(256)
__attribute__((amdgpu_waves_per_eu(2)))
void bahdanau_gemm_logits(
    const float* __restrict__ features, const float* __restrict__ W1,
    const float* __restrict__ W1b, const float* __restrict__ projh,
    const float* __restrict__ Vw, const float* __restrict__ Vb,
    float* __restrict__ logits)
{
    extern __shared__ unsigned short smem[];
    unsigned short* At  = smem;                    // [row][k], row stride 1024
    unsigned short* Bt0 = smem + 128 * 1024;       // [n][k], row stride 32
    unsigned short* Bt1 = Bt0 + 128 * 32;

    const int t     = threadIdx.x;
    const int lane  = t & 31;          // wave32
    const int wid   = t >> 5;          // 8 waves
    const int half  = lane >> 4;       // K-half selector per ISA layout
    const int l16   = lane & 15;
    const int mq    = wid & 3;         // row quarter: rows mq*32 .. mq*32+31
    const int nhalf = wid >> 2;        // n half within the 128-col block

    const int r0 = blockIdx.x * 128;   // flattened (b,s) row base
    const int b  = r0 >> 10;           // 128 | 1024 -> one batch per WG

    // ---- Stage the full A panel once: features[r0..r0+127, 0..1023] -> bf16
#pragma unroll 2
    for (int p = 0; p < 128; ++p) {
        int idx = p * 256 + t;             // quad index
        int row = idx >> 8;                // 0..127
        int col = (idx & 255) * 4;         // 0..1020
        const float4 v = *(const float4*)(features + (size_t)(r0 + row) * F_ + col);
        uint2 pk;
        pk.x = pack_bf16x2(v.x, v.y);
        pk.y = pack_bf16x2(v.z, v.w);
        *(uint2*)(At + row * 1024 + col) = pk;
    }

    // W1 tile stager: [k0..k0+31] x [nb..nb+127] -> LDS transposed [n][k].
    auto stageB = [&](unsigned short* dst, int nb, int k0) {
#pragma unroll
        for (int p = 0; p < 8; ++p) {
            int idx = p * 256 + t;         // pair index, 0..2047
            int kk  = (idx >> 7) * 2;      // 0,2,..,30
            int n   = idx & 127;
            float v0 = W1[(size_t)(k0 + kk)     * U_ + (nb + n)];
            float v1 = W1[(size_t)(k0 + kk + 1) * U_ + (nb + n)];
            *(unsigned int*)(dst + n * 32 + kk) = pack_bf16x2(v0, v1);
        }
    };

    // Per-lane logit partials: 2 M-frags x 8 C-rows = this wave's 32 rows
    // (over its n-half only; combined across n-halves at the end).
    float lp[16];
#pragma unroll
    for (int j = 0; j < 16; ++j) lp[j] = 0.f;

    for (int nb = 0; nb < U_; nb += 128) {
        v8f acc[8];                        // [nf*2 + mf]
#pragma unroll
        for (int q = 0; q < 8; ++q)
#pragma unroll
            for (int j = 0; j < 8; ++j) acc[q][j] = 0.f;

        stageB(Bt0, nb, 0);
        __syncthreads();                   // A panel + first B tile visible

        for (int k0 = 0; k0 < F_; k0 += 32) {
            unsigned short* Bcur = ((k0 >> 5) & 1) ? Bt1 : Bt0;
            unsigned short* Bnxt = ((k0 >> 5) & 1) ? Bt0 : Bt1;

            // Stage next tile (uniform branch; overlaps the WMMAs below).
            if (k0 + 32 < F_) stageB(Bnxt, nb, k0 + 32);
            __builtin_prefetch(
                &W1[(size_t)((k0 + 64) & (F_ - 1)) * U_ + nb + (t & 127)], 0, 1);

            // Two A fragments (rows mq*32..+15 and +16..+31) from the panel.
            FragU a0, a1;
            const unsigned short* Ar = At + (mq * 32 + l16) * 1024 + k0 + half * 8;
            a0.q.lo = *(const uint4*)(Ar);
            a0.q.hi = *(const uint4*)(Ar + 16);
            a1.q.lo = *(const uint4*)(Ar + 16 * 1024);
            a1.q.hi = *(const uint4*)(Ar + 16 * 1024 + 16);

            // 4 B fragments, each consumed by two WMMAs (a0 and a1).
#pragma unroll
            for (int nf = 0; nf < 4; ++nf) {
                FragU bb;
                const unsigned short* pn =
                    Bcur + (nhalf * 64 + nf * 16 + l16) * 32 + half * 8;
                bb.q.lo = *(const uint4*)pn;
                bb.q.hi = *(const uint4*)(pn + 16);
                acc[nf * 2 + 0] = __builtin_amdgcn_wmma_f32_16x16x32_bf16(
                    false, a0.v, false, bb.v, (short)0, acc[nf * 2 + 0], false, false);
                acc[nf * 2 + 1] = __builtin_amdgcn_wmma_f32_16x16x32_bf16(
                    false, a1.v, false, bb.v, (short)0, acc[nf * 2 + 1], false, false);
            }

            __syncthreads();               // one barrier per k-step (double buffer)
        }

        // Fused epilogue: bias + broadcast + hw-tanh + V-dot over this n-half.
        // C layout: VGPR j -> row j (lanes 0-15) / row j+8 (lanes 16-31), N = l16.
#pragma unroll
        for (int nf = 0; nf < 4; ++nf) {
            const int n      = nb + nhalf * 64 + nf * 16 + l16;
            const float bias = W1b[n];
            const float ph   = projh[b * U_ + n];
            const float vw   = Vw[n];
#pragma unroll
            for (int mf = 0; mf < 2; ++mf)
#pragma unroll
                for (int j = 0; j < 8; ++j)
                    lp[mf * 8 + j] +=
                        fast_tanh(acc[nf * 2 + mf][j] + bias + ph) * vw;
        }
    }

    // Reduce partials across the 16 lanes holding the same row.
#pragma unroll
    for (int j = 0; j < 16; ++j) {
        float v = lp[j];
        v += __shfl_xor(v, 8, 32);
        v += __shfl_xor(v, 4, 32);
        v += __shfl_xor(v, 2, 32);
        v += __shfl_xor(v, 1, 32);
        lp[j] = v;
    }

    // Combine the two n-halves via LDS (reuse B-tile region; WMMA work done).
    float* lred = (float*)Bt0;             // 2 x 128 floats
    __syncthreads();
    if (l16 == 0) {
#pragma unroll
        for (int mf = 0; mf < 2; ++mf)
#pragma unroll
            for (int j = 0; j < 8; ++j) {
                int row = mq * 32 + mf * 16 + half * 8 + j;
                lred[nhalf * 128 + row] = lp[mf * 8 + j];
            }
    }
    __syncthreads();
    if (t < 128)
        logits[r0 + t] = lred[t] + lred[128 + t] + Vb[0];
}

// ---------------------------------------------------------------------------
// Kernel C: softmax over the sequence axis, one workgroup per batch row.
// ---------------------------------------------------------------------------
__global__ __launch_bounds__(256) void softmax_kernel(
    float* __restrict__ logits_weights, float* __restrict__ out_weights)
{
    __shared__ float red[256];
    const int b = blockIdx.x;
    const int t = threadIdx.x;
    float* lrow = logits_weights + b * S_;

    float x0 = lrow[t], x1 = lrow[t + 256], x2 = lrow[t + 512], x3 = lrow[t + 768];
    float m = fmaxf(fmaxf(x0, x1), fmaxf(x2, x3));
    red[t] = m; __syncthreads();
    for (int s = 128; s > 0; s >>= 1) {
        if (t < s) red[t] = fmaxf(red[t], red[t + s]);
        __syncthreads();
    }
    m = red[0]; __syncthreads();

    float e0 = __expf(x0 - m), e1 = __expf(x1 - m),
          e2 = __expf(x2 - m), e3 = __expf(x3 - m);
    red[t] = e0 + e1 + e2 + e3; __syncthreads();
    for (int s = 128; s > 0; s >>= 1) {
        if (t < s) red[t] += red[t + s];
        __syncthreads();
    }
    const float inv = 1.f / red[0];

    float w0 = e0 * inv, w1 = e1 * inv, w2 = e2 * inv, w3 = e3 * inv;
    lrow[t] = w0; lrow[t + 256] = w1; lrow[t + 512] = w2; lrow[t + 768] = w3;
    float* orow = out_weights + b * S_;
    orow[t] = w0; orow[t + 256] = w1; orow[t + 512] = w2; orow[t + 768] = w3;
}

// ---------------------------------------------------------------------------
// Kernel D: context[b,f] = sum_s w[b,s] * features[b,s,f]  (memory bound)
// ---------------------------------------------------------------------------
__global__ __launch_bounds__(256) void context_kernel(
    const float* __restrict__ features, const float* __restrict__ weights,
    float* __restrict__ ctx)
{
    __shared__ float wsh[S_];
    const int b = blockIdx.x, chunk = blockIdx.y, t = threadIdx.x;
#pragma unroll
    for (int p = 0; p < 4; ++p)
        wsh[p * 256 + t] = weights[b * S_ + p * 256 + t];
    __syncthreads();

    const int f = chunk * 256 + t;
    const float* fb = features + (size_t)b * S_ * F_ + f;
    float acc = 0.f;
#pragma unroll 4
    for (int s = 0; s < S_; ++s)
        acc += wsh[s] * fb[(size_t)s * F_];
    ctx[b * F_ + f] = acc;
}

// ---------------------------------------------------------------------------
extern "C" void kernel_launch(void* const* d_in, const int* in_sizes, int n_in,
                              void* d_out, int out_size, void* d_ws, size_t ws_size,
                              hipStream_t stream) {
    const float* features = (const float*)d_in[0];  // [64,1024,1024]
    const float* hidden   = (const float*)d_in[1];  // [64,1024]
    const float* W1w      = (const float*)d_in[2];  // [1024,1024]
    const float* W1b      = (const float*)d_in[3];  // [1024]
    const float* W2w      = (const float*)d_in[4];  // [1024,1024]
    const float* W2b      = (const float*)d_in[5];  // [1024]
    const float* Vw       = (const float*)d_in[6];  // [1024,1]
    const float* Vb       = (const float*)d_in[7];  // [1]

    float* out  = (float*)d_out;
    float* ctx  = out;             // context_vector [64,1024]
    float* attw = out + B_ * S_;   // attention_weights [64,1024,1]

    float* wsf    = (float*)d_ws;
    float* projh  = wsf;            // 64*1024 f32
    float* logits = wsf + B_ * U_;  // 64*1024 f32 (becomes weights in-place)

    // 256 KB A panel + 2x8 KB W1 tiles of dynamic LDS (CDNA5: 320 KB per WG).
    const int shmem_bytes = (128 * 1024 + 2 * 128 * 32) * (int)sizeof(unsigned short);
    hipFuncSetAttribute((const void*)bahdanau_gemm_logits,
                        hipFuncAttributeMaxDynamicSharedMemorySize, shmem_bytes);

    proj_hidden_kernel<<<dim3(4, 64), 256, 0, stream>>>(hidden, W2w, W2b, projh);
    bahdanau_gemm_logits<<<dim3(512), 256, shmem_bytes, stream>>>(
        features, W1w, W1b, projh, Vw, Vb, logits);
    softmax_kernel<<<dim3(64), 256, 0, stream>>>(logits, attw);
    context_kernel<<<dim3(64, 4), 256, 0, stream>>>(features, logits, ctx);
}
